// bingo_cc_31653908971618
// MI455X (gfx1250) — matrix operand order repaired and checked by reading, not compiled, not run on hardware
//
#include <hip/hip_runtime.h>
#include <hip/hip_bf16.h>

// ============================================================================
// CCNet-style head for MI455X (gfx1250, wave32, WMMA).
// All convolutions are implicit-GEMM on v_wmma_f32_16x16x32_bf16:
//   A (16x32)  = pre-swizzled bf16 weight fragments (ISA 7.12.2 layout)
//   B (32x16)  = 32 input channels x 16 consecutive w-positions (contiguous)
//   C (16x16)  = f32 accumulator, 8 VGPRs
// Each wave owns 64 output channels x 48 w-positions (4x3 accumulators):
// one set of 4 A fragments feeds 12 WMMAs, tripling A-fragment reuse; the
// 3 B fragments come from one contiguous activation row segment. The K loop
// is taps x channel-chunks with pointer-increment-only addressing.
// Activations are NCHW bf16 with a 1-pixel zero border so every 3x3 tap is a
// contiguous row load. Stats (BN/GN) are f32 block reductions.
// ============================================================================

typedef __attribute__((ext_vector_type(16))) __bf16 v16bf;
typedef __attribute__((ext_vector_type(8)))  float  v8f;

// ---------------------------------------------------------------------------
// zero a buffer (u32 granularity, grid-stride)
// ---------------------------------------------------------------------------
__global__ void zero_u32_kernel(uint32_t* __restrict__ p, size_t n) {
  size_t i = (size_t)blockIdx.x * blockDim.x + threadIdx.x;
  size_t stride = (size_t)gridDim.x * blockDim.x;
  for (; i < n; i += stride) p[i] = 0u;
}

// ---------------------------------------------------------------------------
// x (N,C,H,W) f32  ->  xb (N,C,H+2,W+2) bf16 with zero border
// ---------------------------------------------------------------------------
__global__ void pad_x_kernel(const float* __restrict__ x, __bf16* __restrict__ xb,
                             int NC, int H, int W, int Hp, int Wp) {
  size_t idx = (size_t)blockIdx.x * blockDim.x + threadIdx.x;
  size_t total = (size_t)NC * Hp * Wp;
  if (idx >= total) return;
  int wp = (int)(idx % Wp);
  size_t t = idx / Wp;
  int hp = (int)(t % Hp);
  size_t nc = t / Hp;
  int h = hp - 1, w = wp - 1;
  float v = 0.f;
  if (h >= 0 && h < H && w >= 0 && w < W)
    v = x[nc * H * W + (size_t)h * W + w];
  xb[idx] = (__bf16)v;
}

// ---------------------------------------------------------------------------
// Weight pre-swizzle into WMMA A-fragment order, grouped so the 4 o_tiles a
// wave consumes per K-step are contiguous:
//   dst layout: [o_group][kstep][sub(4)][lane(32)][e(16)]   (bf16)
// K ordering: K = (kh*KW+kw)*Cin + ci (chunks of 32 never straddle a tap:
// Cin % 32 == 0 for every conv here).
// ISA 16-bit A 16x32 per-lane map: K_local = (e&7) + (e>=8?16:0) + (lane>=16?8:0)
// ---------------------------------------------------------------------------
__global__ void prep_weights_kernel(const float* __restrict__ src, __bf16* __restrict__ dst,
                                    int O, int Opad, int Cin, int KHW, int ksteps) {
  size_t idx = (size_t)blockIdx.x * blockDim.x + threadIdx.x;
  size_t total = (size_t)Opad * KHW * Cin;   // == (Opad/64) * ksteps * 2048
  if (idx >= total) return;
  int e    = (int)(idx & 15);
  int lane = (int)((idx >> 4) & 31);
  int sub  = (int)((idx >> 9) & 3);
  size_t rest = idx >> 11;
  int ks = (int)(rest % (size_t)ksteps);
  int og = (int)(rest / (size_t)ksteps);
  int Klocal = (e & 7) + ((e >> 3) << 4) + (((lane >> 4) & 1) << 3);
  int K = ks * 32 + Klocal;
  int khkw = K / Cin;
  int ci = K - khkw * Cin;
  int o = (og * 4 + sub) * 16 + (lane & 15);
  float v = 0.f;
  if (o < O) v = src[((size_t)o * Cin + ci) * KHW + khkw];
  dst[idx] = (__bf16)v;
}

// ---------------------------------------------------------------------------
// Implicit-GEMM conv, bf16 WMMA, f32 accumulate.
// One wave = 64 output channels x 48 w-positions (4x3 WMMA accumulators).
// Flat wave-indexed grid: wid -> (n, h, w-tile-group, co-group); 8 waves/block.
// Dual input (in0 first C0 channels, in1 the rest) supports the virtual
// concat conv without materializing the 2560-channel tensor.
// yoff/xoff = buffer_border - conv_pad.
// ---------------------------------------------------------------------------
__global__ void __launch_bounds__(256)
conv_wmma_kernel(const __bf16* __restrict__ in0, int C0,
                 const __bf16* __restrict__ in1, int C1,
                 int Hp, int Wp, int yoff, int xoff,
                 const __bf16* __restrict__ afrag,
                 const float* __restrict__ bias,
                 float* __restrict__ out,
                 int N, int Cout, int H, int W, int KHW,
                 int co_groups) {
  const int lane = threadIdx.x & 31;
  const int wid  = blockIdx.x * 8 + (threadIdx.x >> 5);
  const int Wg = W / 48;                    // w-tile groups (3 tiles each)
  int t = wid;
  const int cg  = t % co_groups; t /= co_groups;
  const int wtg = t % Wg;        t /= Wg;
  const int h   = t % H;
  const int n   = t / H;
  if (n >= N) return;

  const int w0 = wtg * 48;                  // tiles at w0, w0+16, w0+32
  const int chunks0 = C0 >> 5;
  const int chunks1 = C1 >> 5;
  const int ksteps  = KHW * (chunks0 + chunks1);
  const size_t cstride = (size_t)32 * Hp * Wp;   // +32 input channels

  const __bf16* aptr = afrag + ((size_t)cg * ksteps) * 2048 + (size_t)lane * 16;

  v8f acc[4][3] = {};

  for (int tap = 0; tap < KHW; ++tap) {
    const int kh = tap / 3;                 // tap==0 for 1x1 convs
    const int kw = tap - kh * 3;
    const int row = h + kh + yoff;
    const int col = w0 + kw + xoff;

    // ---- channels from in0 ----
    const __bf16* bptr = in0 + (((size_t)n * C0) * Hp + row) * Wp + col + (lane & 15);
    for (int c = 0; c < chunks0; ++c) {
      if (c + 1 < chunks0) __builtin_prefetch(bptr + cstride, 0, 1);
      __builtin_prefetch(aptr + 2048, 0, 1);     // A stream: one chunk ahead
      v16bf b[3];
#pragma unroll
      for (int u = 0; u < 3; ++u)
#pragma unroll
        for (int e = 0; e < 16; ++e)
          b[u][e] = bptr[(size_t)((e & 7) + ((e >> 3) << 4) + (((lane >> 4) & 1) << 3)) * Hp * Wp + u * 16];
      bptr += cstride;
      v16bf a[4];
#pragma unroll
      for (int s = 0; s < 4; ++s) a[s] = *(const v16bf*)(aptr + (size_t)s * 512);
      aptr += 2048;
#pragma unroll
      for (int s = 0; s < 4; ++s)
#pragma unroll
        for (int u = 0; u < 3; ++u)
          acc[s][u] = __builtin_amdgcn_wmma_f32_16x16x32_bf16(
              false, a[s], false, b[u], (short)0, acc[s][u], false, false);
    }

    // ---- channels from in1 (virtual concat) ----
    if (chunks1) {
      const __bf16* bptr1 = in1 + (((size_t)n * C1) * Hp + row) * Wp + col + (lane & 15);
      for (int c = 0; c < chunks1; ++c) {
        if (c + 1 < chunks1) __builtin_prefetch(bptr1 + cstride, 0, 1);
        __builtin_prefetch(aptr + 2048, 0, 1);
        v16bf b[3];
#pragma unroll
        for (int u = 0; u < 3; ++u)
#pragma unroll
          for (int e = 0; e < 16; ++e)
            b[u][e] = bptr1[(size_t)((e & 7) + ((e >> 3) << 4) + (((lane >> 4) & 1) << 3)) * Hp * Wp + u * 16];
        bptr1 += cstride;
        v16bf a[4];
#pragma unroll
        for (int s = 0; s < 4; ++s) a[s] = *(const v16bf*)(aptr + (size_t)s * 512);
        aptr += 2048;
#pragma unroll
        for (int s = 0; s < 4; ++s)
#pragma unroll
          for (int u = 0; u < 3; ++u)
            acc[s][u] = __builtin_amdgcn_wmma_f32_16x16x32_bf16(
                false, a[s], false, b[u], (short)0, acc[s][u], false, false);
      }
    }
  }

  // C layout: VGPR r, lanes 0-15 -> M=r ; lanes 16-31 -> M=8+r ; N=lane&15
  const int m_hi = ((lane >> 4) & 1) << 3;
  const int wl = lane & 15;
#pragma unroll
  for (int s = 0; s < 4; ++s) {
#pragma unroll
    for (int u = 0; u < 3; ++u) {
      const int w = w0 + u * 16 + wl;
#pragma unroll
      for (int r = 0; r < 8; ++r) {
        int co = (cg * 4 + s) * 16 + m_hi + r;
        if (co < Cout) {
          float v = acc[s][u][r];
          if (bias) v += bias[co];
          out[(((size_t)n * Cout + co) * H + h) * W + w] = v;
        }
      }
    }
  }
}

// ---------------------------------------------------------------------------
// BatchNorm stats: one block per channel, LDS tree reduction (deterministic).
// ---------------------------------------------------------------------------
__global__ void __launch_bounds__(256)
bn_stats_kernel(const float* __restrict__ y, float* __restrict__ mean,
                float* __restrict__ rstd, int N, int C, int HW) {
  __shared__ float s_sum[256];
  __shared__ float s_sq[256];
  const int c = blockIdx.x;
  const int total = N * HW;
  float sum = 0.f, sq = 0.f;
  for (int i = threadIdx.x; i < total; i += 256) {
    int nn = i / HW;
    int r  = i - nn * HW;
    float v = y[((size_t)nn * C + c) * HW + r];
    sum += v; sq += v * v;
  }
  s_sum[threadIdx.x] = sum; s_sq[threadIdx.x] = sq;
  __syncthreads();
  for (int off = 128; off > 0; off >>= 1) {
    if ((int)threadIdx.x < off) {
      s_sum[threadIdx.x] += s_sum[threadIdx.x + off];
      s_sq[threadIdx.x]  += s_sq[threadIdx.x + off];
    }
    __syncthreads();
  }
  if (threadIdx.x == 0) {
    float m = s_sum[0] / (float)total;
    float v = s_sq[0] / (float)total - m * m;
    mean[c] = m;
    rstd[c] = rsqrtf(v + 1e-5f);
  }
}

// GroupNorm stats: one block per group (batch==1 here).
__global__ void __launch_bounds__(256)
gn_stats_kernel(const float* __restrict__ y, float* __restrict__ mean,
                float* __restrict__ rstd, int Cg, int HW) {
  __shared__ float s_sum[256];
  __shared__ float s_sq[256];
  const int g = blockIdx.x;
  const size_t total = (size_t)Cg * HW;
  float sum = 0.f, sq = 0.f;
  for (size_t i = threadIdx.x; i < total; i += 256) {
    int cl = (int)(i / (size_t)HW);
    int r  = (int)(i - (size_t)cl * HW);
    float v = y[((size_t)(g * Cg + cl)) * HW + r];
    sum += v; sq += v * v;
  }
  s_sum[threadIdx.x] = sum; s_sq[threadIdx.x] = sq;
  __syncthreads();
  for (int off = 128; off > 0; off >>= 1) {
    if ((int)threadIdx.x < off) {
      s_sum[threadIdx.x] += s_sum[threadIdx.x + off];
      s_sq[threadIdx.x]  += s_sq[threadIdx.x + off];
    }
    __syncthreads();
  }
  if (threadIdx.x == 0) {
    float m = s_sum[0] / (float)total;
    float v = s_sq[0] / (float)total - m * m;
    mean[g] = m;
    rstd[g] = rsqrtf(v + 1e-5f);
  }
}

// ---------------------------------------------------------------------------
// normalize + ReLU + bf16, writing interior of a padded (Hp,Wp) buffer
// ---------------------------------------------------------------------------
__global__ void bn_apply_pad_kernel(const float* __restrict__ y,
                                    const float* __restrict__ mean,
                                    const float* __restrict__ rstd,
                                    __bf16* __restrict__ dst,
                                    int N, int C, int H, int W, int Hp, int Wp) {
  size_t idx = (size_t)blockIdx.x * blockDim.x + threadIdx.x;
  size_t total = (size_t)N * C * H * W;
  if (idx >= total) return;
  int w = (int)(idx % W);
  size_t t = idx / W;
  int h = (int)(t % H); t /= H;
  int c = (int)(t % C);
  size_t n = t / C;
  float v = (y[idx] - mean[c]) * rstd[c];
  v = fmaxf(v, 0.f);
  dst[((n * C + c) * Hp + (h + 1)) * Wp + (w + 1)] = (__bf16)v;
}

// normalize + ReLU + bf16, scattering into the "bingo" 4x4 mosaic (512,194,194)
__global__ void bn_apply_bingo_kernel(const float* __restrict__ y,
                                      const float* __restrict__ mean,
                                      const float* __restrict__ rstd,
                                      __bf16* __restrict__ big) {
  size_t idx = (size_t)blockIdx.x * blockDim.x + threadIdx.x;
  const size_t total = (size_t)16 * 512 * 48 * 48;
  if (idx >= total) return;
  int w = (int)(idx % 48);
  size_t t = idx / 48;
  int h = (int)(t % 48); t /= 48;
  int c = (int)(t % 512);
  int n = (int)(t / 512);
  float v = (y[idx] - mean[c]) * rstd[c];
  v = fmaxf(v, 0.f);
  int gy = (n >> 2) * 48 + h + 1;   // +1: border of padded big buffer
  int gx = (n & 3) * 48 + w + 1;
  big[((size_t)c * 194 + gy) * 194 + gx] = (__bf16)v;
}

// GN normalize + ReLU + bf16, gathering mosaic back into (16,256,48,48)
__global__ void gn_apply_post_kernel(const float* __restrict__ yb,
                                     const float* __restrict__ mean,
                                     const float* __restrict__ rstd,
                                     __bf16* __restrict__ post) {
  size_t idx = (size_t)blockIdx.x * blockDim.x + threadIdx.x;
  const size_t total = (size_t)16 * 256 * 48 * 48;
  if (idx >= total) return;
  int w = (int)(idx % 48);
  size_t t = idx / 48;
  int h = (int)(t % 48); t /= 48;
  int c = (int)(t % 256);
  int n = (int)(t / 256);
  float s = yb[((size_t)c * 192 + (n >> 2) * 48 + h) * 192 + (n & 3) * 48 + w];
  int g = c >> 4;
  float v = fmaxf((s - mean[g]) * rstd[g], 0.f);
  post[idx] = (__bf16)v;
}

// ---------------------------------------------------------------------------
// Criss-cross attention energies: e[b,h,w,s], s<H = column attn (masked diag),
// s>=H = row attn. Dot over C=64 channels.
// ---------------------------------------------------------------------------
__global__ void energy_kernel(const float* __restrict__ q, const float* __restrict__ k,
                              float* __restrict__ e, int B, int C, int H, int W) {
  const int S = H + W;
  size_t idx = (size_t)blockIdx.x * blockDim.x + threadIdx.x;
  size_t total = (size_t)B * H * W * S;
  if (idx >= total) return;
  int s = (int)(idx % S);
  size_t t = idx / S;
  int w = (int)(t % W); t /= W;
  int h = (int)(t % H);
  int b = (int)(t / H);
  if (s < H && s == h) { e[idx] = -__builtin_inff(); return; }
  const int HW = H * W;
  const float* qp = q + (size_t)b * C * HW + h * W + w;
  const float* kp = k + (size_t)b * C * HW +
                    ((s < H) ? (s * W + w) : (h * W + (s - H)));
  float acc = 0.f;
  for (int c = 0; c < C; ++c) acc += qp[(size_t)c * HW] * kp[(size_t)c * HW];
  e[idx] = acc;
}

// in-place softmax over last dim S (=96)
__global__ void softmax_kernel(float* __restrict__ e, int rows, int S) {
  int r = blockIdx.x * blockDim.x + threadIdx.x;
  if (r >= rows) return;
  float* p = e + (size_t)r * S;
  float m = -__builtin_inff();
  for (int s = 0; s < S; ++s) m = fmaxf(m, p[s]);
  float sum = 0.f;
  for (int s = 0; s < S; ++s) { float v = __expf(p[s] - m); p[s] = v; sum += v; }
  float inv = 1.f / sum;
  for (int s = 0; s < S; ++s) p[s] *= inv;
}

// aggregation + residual: out = gamma*(outH + outW) + x, written into padded bf16
__global__ void cca_out_kernel(const float* __restrict__ v, const float* __restrict__ att,
                               const __bf16* __restrict__ fin, __bf16* __restrict__ fout,
                               const float* __restrict__ gamma_p,
                               int B, int C, int H, int W, int Hp, int Wp) {
  size_t idx = (size_t)blockIdx.x * blockDim.x + threadIdx.x;
  size_t total = (size_t)B * C * H * W;
  if (idx >= total) return;
  int w = (int)(idx % W);
  size_t t = idx / W;
  int h = (int)(t % H); t /= H;
  int c = (int)(t % C);
  int b = (int)(t / C);
  const float* vb = v + ((size_t)b * C + c) * H * W;
  const float* at = att + (((size_t)b * H + h) * W + w) * (H + W);
  float acc = 0.f;
  for (int s = 0; s < H; ++s) acc += vb[(size_t)s * W + w] * at[s];
  for (int s = 0; s < W; ++s) acc += vb[(size_t)h * W + s] * at[H + s];
  size_t pidx = (((size_t)b * C + c) * Hp + (h + 1)) * Wp + (w + 1);
  float res = (*gamma_p) * acc + (float)fin[pidx];
  fout[pidx] = (__bf16)res;
}

// ============================================================================
// Host launch
// ============================================================================
static inline unsigned blk(size_t total, int t = 256) {
  return (unsigned)((total + t - 1) / t);
}

extern "C" void kernel_launch(void* const* d_in, const int* in_sizes, int n_in,
                              void* d_out, int out_size, void* d_ws, size_t ws_size,
                              hipStream_t stream) {
  (void)in_sizes; (void)n_in; (void)out_size; (void)ws_size;
  const float* x       = (const float*)d_in[0];
  const float* conva_w = (const float*)d_in[1];
  const float* wq      = (const float*)d_in[2];
  const float* bq      = (const float*)d_in[3];
  const float* wk      = (const float*)d_in[4];
  const float* bk      = (const float*)d_in[5];
  const float* wv      = (const float*)d_in[6];
  const float* bv      = (const float*)d_in[7];
  const float* gamma   = (const float*)d_in[8];
  const float* convb_w = (const float*)d_in[9];
  const float* cat_w   = (const float*)d_in[10];
  const float* bingo_w = (const float*)d_in[11];
  const float* cls_w   = (const float*)d_in[12];
  const float* cls_b   = (const float*)d_in[13];
  float* out = (float*)d_out;

  const int N = 16, H = 48, W = 48, HP = 50, WP = 50;
  const int HW = H * W;
  const int BH = 192, BHP = 194;

  char* ws = (char*)d_ws;
  size_t off = 0;
  auto alloc = [&](size_t bytes) -> char* {
    char* p = ws + off;
    off += (bytes + 255) & ~(size_t)255;
    return p;
  };

  __bf16* xb    = (__bf16*)alloc((size_t)N * 2048 * HP * WP * 2);
  __bf16* afrA  = (__bf16*)alloc((size_t)512 * 9 * 2048 * 2);
  __bf16* afrQ  = (__bf16*)alloc((size_t)64 * 512 * 2);
  __bf16* afrK  = (__bf16*)alloc((size_t)64 * 512 * 2);
  __bf16* afrV  = (__bf16*)alloc((size_t)512 * 512 * 2);
  __bf16* afrB  = (__bf16*)alloc((size_t)512 * 9 * 512 * 2);
  __bf16* afrC  = (__bf16*)alloc((size_t)512 * 9 * 2560 * 2);
  __bf16* afrG  = (__bf16*)alloc((size_t)256 * 9 * 512 * 2);
  __bf16* afrS  = (__bf16*)alloc((size_t)64 * 256 * 2);
  float*  ybuf  = (float*)alloc((size_t)N * 512 * HW * 4);   // reused f32 conv out
  __bf16* featA = (__bf16*)alloc((size_t)N * 512 * HP * WP * 2);
  __bf16* featB = (__bf16*)alloc((size_t)N * 512 * HP * WP * 2);
  float*  qbuf  = (float*)alloc((size_t)N * 64 * HW * 4);
  float*  kbuf  = (float*)alloc((size_t)N * 64 * HW * 4);
  float*  ebuf  = (float*)alloc((size_t)N * HW * (H + W) * 4);
  __bf16* bigp  = (__bf16*)alloc((size_t)512 * BHP * BHP * 2);
  __bf16* post  = (__bf16*)alloc((size_t)N * 256 * HW * 2);
  float*  meanb = (float*)alloc(4096);
  float*  rstdb = (float*)alloc(4096);
  float*  vbuf  = ybuf;  // v-conv output shares the f32 staging buffer

  // ---- zero bordered bf16 buffers (interiors get overwritten each call) ----
  zero_u32_kernel<<<4096, 256, 0, stream>>>((uint32_t*)featA, (size_t)N * 512 * HP * WP / 2);
  zero_u32_kernel<<<4096, 256, 0, stream>>>((uint32_t*)featB, (size_t)N * 512 * HP * WP / 2);
  zero_u32_kernel<<<4096, 256, 0, stream>>>((uint32_t*)bigp, (size_t)512 * BHP * BHP / 2);

  // ---- convert + pad x ----
  {
    size_t tot = (size_t)N * 2048 * HP * WP;
    pad_x_kernel<<<blk(tot), 256, 0, stream>>>(x, xb, N * 2048, H, W, HP, WP);
  }

  // ---- weight pre-swizzle ----
  auto prep = [&](const float* src, __bf16* dst, int O, int Opad, int Ci, int KHW) {
    int ksteps = KHW * Ci / 32;
    size_t tot = (size_t)Opad * KHW * Ci;
    prep_weights_kernel<<<blk(tot), 256, 0, stream>>>(src, dst, O, Opad, Ci, KHW, ksteps);
  };
  prep(conva_w, afrA, 512, 512, 2048, 9);
  prep(wq,      afrQ,  64,  64,  512, 1);
  prep(wk,      afrK,  64,  64,  512, 1);
  prep(wv,      afrV, 512, 512,  512, 1);
  prep(convb_w, afrB, 512, 512,  512, 9);
  prep(cat_w,   afrC, 512, 512, 2560, 9);
  prep(bingo_w, afrG, 256, 256,  512, 9);
  prep(cls_w,   afrS,  21,  64,  256, 1);

  // ---- conv launcher (flat wave grid: 8 waves / 256-thread block) ----
  auto conv = [&](const __bf16* in0, int C0, const __bf16* in1, int C1,
                  int Hp, int Wp, int yoff, int xoff,
                  const __bf16* afrag, const float* bias, float* outp,
                  int n, int Cout, int h, int w, int KHW) {
    int cog = (Cout + 63) / 64;
    size_t waves = (size_t)n * h * (w / 48) * cog;
    unsigned blocks = (unsigned)((waves + 7) / 8);
    conv_wmma_kernel<<<blocks, 256, 0, stream>>>(in0, C0, in1, C1, Hp, Wp, yoff, xoff,
                                                 afrag, bias, outp, n, Cout, h, w,
                                                 KHW, cog);
  };

  // ---- stage 1: conva (3x3, 2048->512) + BN + ReLU ----
  conv(xb, 2048, xb, 0, HP, WP, 0, 0, afrA, nullptr, ybuf, N, 512, H, W, 9);
  bn_stats_kernel<<<512, 256, 0, stream>>>(ybuf, meanb, rstdb, N, 512, HW);
  {
    size_t tot = (size_t)N * 512 * HW;
    bn_apply_pad_kernel<<<blk(tot), 256, 0, stream>>>(ybuf, meanb, rstdb, featA,
                                                      N, 512, H, W, HP, WP);
  }

  // ---- 2x criss-cross attention ----
  const __bf16* cin = featA;
  __bf16* cout = featB;
  for (int rec = 0; rec < 2; ++rec) {
    conv(cin, 512, cin, 0, HP, WP, 1, 1, afrQ, bq, qbuf, N, 64, H, W, 1);
    conv(cin, 512, cin, 0, HP, WP, 1, 1, afrK, bk, kbuf, N, 64, H, W, 1);
    conv(cin, 512, cin, 0, HP, WP, 1, 1, afrV, bv, vbuf, N, 512, H, W, 1);
    {
      size_t tot = (size_t)N * HW * (H + W);
      energy_kernel<<<blk(tot), 256, 0, stream>>>(qbuf, kbuf, ebuf, N, 64, H, W);
    }
    softmax_kernel<<<blk((size_t)N * HW), 256, 0, stream>>>(ebuf, N * HW, H + W);
    {
      size_t tot = (size_t)N * 512 * HW;
      cca_out_kernel<<<blk(tot), 256, 0, stream>>>(vbuf, ebuf, cin, cout, gamma,
                                                   N, 512, H, W, HP, WP);
    }
    const __bf16* tmp = cin; cin = cout; cout = (__bf16*)tmp;
  }
  // result now in featA (cin == featA)

  // ---- convb (3x3, 512->512) + BN + ReLU -> featB ----
  conv(featA, 512, featA, 0, HP, WP, 0, 0, afrB, nullptr, ybuf, N, 512, H, W, 9);
  bn_stats_kernel<<<512, 256, 0, stream>>>(ybuf, meanb, rstdb, N, 512, HW);
  {
    size_t tot = (size_t)N * 512 * HW;
    bn_apply_pad_kernel<<<blk(tot), 256, 0, stream>>>(ybuf, meanb, rstdb, featB,
                                                      N, 512, H, W, HP, WP);
  }

  // ---- cat conv (3x3, [x(2048) | featB(512)] -> 512) + BN + ReLU -> bingo mosaic ----
  conv(xb, 2048, featB, 512, HP, WP, 0, 0, afrC, nullptr, ybuf, N, 512, H, W, 9);
  bn_stats_kernel<<<512, 256, 0, stream>>>(ybuf, meanb, rstdb, N, 512, HW);
  {
    size_t tot = (size_t)N * 512 * HW;
    bn_apply_bingo_kernel<<<blk(tot), 256, 0, stream>>>(ybuf, meanb, rstdb, bigp);
  }

  // ---- bingo conv (3x3, 512->256 on 192x192) + GN(16) + ReLU -> post ----
  conv(bigp, 512, bigp, 0, BHP, BHP, 0, 0, afrG, nullptr, ybuf, 1, 256, BH, BH, 9);
  gn_stats_kernel<<<16, 256, 0, stream>>>(ybuf, meanb, rstdb, 16, BH * BH);
  {
    size_t tot = (size_t)N * 256 * HW;
    gn_apply_post_kernel<<<blk(tot), 256, 0, stream>>>(ybuf, meanb, rstdb, post);
  }

  // ---- classifier (1x1, 256->21, +bias) -> d_out ----
  conv(post, 256, post, 0, H, W, 0, 0, afrS, cls_b, out, N, 21, H, W, 1);
}
